// TrainableVSA_57329223467250
// MI455X (gfx1250) — matrix operand order repaired and checked by reading, not compiled, not run on hardware
//
#include <hip/hip_runtime.h>
#include <hip/hip_bf16.h>

// ---------------------------------------------------------------------------
// TrainableVSA cleanup similarities for MI455X (gfx1250, wave32, WMMA).
//
// sims[b,l,n] = <dec_n[b,l,:], emb_n[n,:]>  => GEMM M=4096, N=32768, K=512.
// 137 GFLOP, 512MB f32 output => store-bandwidth bound (~22us @ 23.3TB/s).
// bf16 operands (fit in 192MB L2), f32 accumulate in v_wmma_f32_16x16x32_bf16.
// Staging: double-buffered LDS; async global->LDS (ASYNCcnt) when available.
// ---------------------------------------------------------------------------

typedef __bf16 v16bf __attribute__((ext_vector_type(16)));
typedef __bf16 v8bf  __attribute__((ext_vector_type(8)));
typedef float  v8f   __attribute__((ext_vector_type(8)));
typedef int    v4i   __attribute__((ext_vector_type(4)));

#define N_SYM   32768
#define N_DIMS  512
#define BATCH   256
#define SEQLEN  16
#define M_TOT   (BATCH * SEQLEN)   // 4096

#if __has_builtin(__builtin_amdgcn_global_load_async_to_lds_b128) && \
    __has_builtin(__builtin_amdgcn_s_wait_asynccnt)
#define USE_ASYNC_LDS 1
#else
#define USE_ASYNC_LDS 0
#endif

#define AS1 __attribute__((address_space(1)))
#define AS3 __attribute__((address_space(3)))

// ---------------- block-wide reduction (256 threads, wave32) ----------------
__device__ __forceinline__ float block_reduce_sum(float v, float* red8) {
#pragma unroll
    for (int m = 16; m > 0; m >>= 1) v += __shfl_xor(v, m, 32);
    const int lane = threadIdx.x & 31;
    const int wave = threadIdx.x >> 5;
    if (lane == 0) red8[wave] = v;
    __syncthreads();
    if (wave == 0) {
        float t = (lane < 8) ? red8[lane] : 0.0f;
#pragma unroll
        for (int m = 4; m > 0; m >>= 1) t += __shfl_xor(t, m, 32);
        if (lane == 0) red8[0] = t;
    }
    __syncthreads();
    float r = red8[0];
    __syncthreads();   // protect red8 reuse by later calls
    return r;
}

// ---------------- 1) codebook normalize -> bf16 -----------------------------
__global__ void __launch_bounds__(256) emb_norm_kernel(const float* __restrict__ emb,
                                                       __bf16* __restrict__ embN) {
    __shared__ float red8[8];
    const int n = blockIdx.x;
    const int t = threadIdx.x;
    const float* row = emb + (size_t)n * N_DIMS;
    float x0 = row[2 * t], x1 = row[2 * t + 1];
    float ss = block_reduce_sum(x0 * x0 + x1 * x1, red8);
    float inv = rsqrtf(ss);
    __bf16* orow = embN + (size_t)n * N_DIMS;
    orow[2 * t]     = (__bf16)(x0 * inv);
    orow[2 * t + 1] = (__bf16)(x1 * inv);
}

// ---------------- 2) encode + per-position decode -> bf16 -------------------
__global__ void __launch_bounds__(256) dec_prep_kernel(const float* __restrict__ emb,
                                                       const float* __restrict__ pos,
                                                       const int* __restrict__ ids,
                                                       __bf16* __restrict__ decn) {
    __shared__ float red8[8];
    __shared__ int sid[SEQLEN];
    const int b = blockIdx.x;
    const int t = threadIdx.x;
    if (t < SEQLEN) sid[t] = ids[b * SEQLEN + t];
    __syncthreads();

    const int d0 = 2 * t;
    float a0 = 0.0f, a1 = 0.0f;
#pragma unroll
    for (int l = 0; l < SEQLEN; ++l) {
        const float* er = emb + (size_t)sid[l] * N_DIMS;
        float p0 = pos[l * N_DIMS + d0];
        float p1 = pos[l * N_DIMS + d0 + 1];
        a0 += er[d0] * p0;
        a1 += er[d0 + 1] * p1;
    }
    float ss  = block_reduce_sum(a0 * a0 + a1 * a1, red8);
    float inv = rsqrtf(ss);
    a0 *= inv; a1 *= inv;

#pragma unroll
    for (int l = 0; l < SEQLEN; ++l) {
        float p0 = pos[l * N_DIMS + d0];
        float p1 = pos[l * N_DIMS + d0 + 1];
        float v0 = a0 * p0, v1 = a1 * p1;
        float s2 = block_reduce_sum(v0 * v0 + v1 * v1, red8);
        float i2 = rsqrtf(s2);
        __bf16* orow = decn + (size_t)(b * SEQLEN + l) * N_DIMS;
        orow[d0]     = (__bf16)(v0 * i2);
        orow[d0 + 1] = (__bf16)(v1 * i2);
    }
}

// ---------------- 3) WMMA GEMM: C[4096,32768] = A[4096,512] * B[32768,512]^T
// Block tile 64(M) x 128(N), K-step 32. 8 waves = 2(M) x 4(N), each wave a
// 32x32 patch = 4 accumulators. Double-buffered LDS: 2 x (4KB A + 8KB B).
#define TM 64
#define TN 128
#define TK 32
#define KSTEPS (N_DIMS / TK)   // 16

__global__ void __launch_bounds__(256) sim_gemm_kernel(const __bf16* __restrict__ A,
                                                       const __bf16* __restrict__ B,
                                                       float* __restrict__ C) {
    __shared__ __align__(32) __bf16 As[2][TM * TK];
    __shared__ __align__(32) __bf16 Bs[2][TN * TK];

    const int t       = threadIdx.x;
    const int lane    = t & 31;
    const int wave    = t >> 5;
    const int lane_lo = lane & 15;   // N column / M row within a 16-tile
    const int lane_hi = lane >> 4;   // selects K/M sub-range per ISA layout

    const int m_blk = blockIdx.y * TM;
    const int n_blk = blockIdx.x * TN;
    const int wm = (wave & 1) * 32;       // wave's M offset in block tile
    const int wn = (wave >> 1) * 32;      // wave's N offset in block tile

    // staging coordinates: A 64x32 (8 halves/thread), B 128x32 (16 halves/thread)
    const int ar = t >> 2, ac = (t & 3) * 8;
    const int br = t >> 1, bc = (t & 1) * 16;
    const __bf16* Ag = A + (size_t)(m_blk + ar) * N_DIMS + ac;
    const __bf16* Bg = B + (size_t)(n_blk + br) * N_DIMS + bc;
    const int a_lds = ar * TK + ac;
    const int b_lds = br * TK + bc;

    v8f c00 = {}, c01 = {}, c10 = {}, c11 = {};

    // stage one K-tile into LDS buffer `buf`
    auto stage = [&](int buf, int k0) {
#if USE_ASYNC_LDS
        __builtin_amdgcn_global_load_async_to_lds_b128(
            (AS1 v4i*)(Ag + k0), (AS3 v4i*)&As[buf][a_lds], 0, 0);
        __builtin_amdgcn_global_load_async_to_lds_b128(
            (AS1 v4i*)(Bg + k0), (AS3 v4i*)&Bs[buf][b_lds], 0, 0);
        __builtin_amdgcn_global_load_async_to_lds_b128(
            (AS1 v4i*)(Bg + k0 + 8), (AS3 v4i*)&Bs[buf][b_lds + 8], 0, 0);
#else
        *(v8bf*)&As[buf][a_lds]  = *(const v8bf*)(Ag + k0);
        *(v16bf*)&Bs[buf][b_lds] = *(const v16bf*)(Bg + k0);
#endif
    };

    // prologue: fill buffer 0
    stage(0, 0);
#if USE_ASYNC_LDS
    __builtin_amdgcn_s_wait_asynccnt(0);
#endif
    __syncthreads();

    for (int it = 0; it < KSTEPS; ++it) {
        const int cur = it & 1;
        const int nxt = cur ^ 1;
        if (it + 1 < KSTEPS) {
            stage(nxt, (it + 1) * TK);                 // overlap with compute below
            if (it + 2 < KSTEPS) {                     // warm L2 for tile it+2
                __builtin_prefetch(Ag + (it + 2) * TK, 0, 1);
                __builtin_prefetch(Bg + (it + 2) * TK, 0, 1);
            }
        }

        // --- A fragments (16-bit A 16x32): lane row = lane_lo,
        // halves j = K[16*lane_hi + j] (same K order as B)
        v16bf a0, a1;
        {
            const __bf16* base0 = &As[cur][(wm + lane_lo) * TK + 16 * lane_hi];
            const __bf16* base1 = &As[cur][(wm + 16 + lane_lo) * TK + 16 * lane_hi];
            v8bf l0 = *(const v8bf*)(base0), h0 = *(const v8bf*)(base0 + 8);
            v8bf l1 = *(const v8bf*)(base1), h1 = *(const v8bf*)(base1 + 8);
#pragma unroll
            for (int i = 0; i < 8; ++i) {
                a0[i] = l0[i]; a0[8 + i] = h0[i];
                a1[i] = l1[i]; a1[8 + i] = h1[i];
            }
        }
        // --- B fragments (32x16): lane col = lane_lo, halves j = K[16*lane_hi + j]
        v16bf b0 = *(const v16bf*)&Bs[cur][(wn + lane_lo) * TK + 16 * lane_hi];
        v16bf b1 = *(const v16bf*)&Bs[cur][(wn + 16 + lane_lo) * TK + 16 * lane_hi];

        c00 = __builtin_amdgcn_wmma_f32_16x16x32_bf16(false, a0, false, b0, (short)0, c00, false, false);
        c01 = __builtin_amdgcn_wmma_f32_16x16x32_bf16(false, a0, false, b1, (short)0, c01, false, false);
        c10 = __builtin_amdgcn_wmma_f32_16x16x32_bf16(false, a1, false, b0, (short)0, c10, false, false);
        c11 = __builtin_amdgcn_wmma_f32_16x16x32_bf16(false, a1, false, b1, (short)0, c11, false, false);

#if USE_ASYNC_LDS
        if (it + 1 < KSTEPS) __builtin_amdgcn_s_wait_asynccnt(0);
#endif
        __syncthreads();   // nxt fully staged; cur free for overwrite next iter
    }

    // --- store C (f32 16x16 layout: VGPR r holds row r + 8*lane_hi, col lane_lo)
    float* Cp = C + (size_t)(m_blk + wm) * (size_t)N_SYM + (size_t)(n_blk + wn);
#pragma unroll
    for (int r = 0; r < 8; ++r) {
        const int m = r + 8 * lane_hi;
        Cp[(size_t)m * N_SYM + lane_lo]             = c00[r];
        Cp[(size_t)m * N_SYM + 16 + lane_lo]        = c01[r];
        Cp[(size_t)(m + 16) * N_SYM + lane_lo]      = c10[r];
        Cp[(size_t)(m + 16) * N_SYM + 16 + lane_lo] = c11[r];
    }
}

// ---------------------------------------------------------------------------
extern "C" void kernel_launch(void* const* d_in, const int* in_sizes, int n_in,
                              void* d_out, int out_size, void* d_ws, size_t ws_size,
                              hipStream_t stream) {
    const float* emb = (const float*)d_in[0];   // [32768, 512] f32
    const float* pos = (const float*)d_in[1];   // [16, 512]    f32
    const int*   ids = (const int*)d_in[2];     // [256, 16]    i32
    float*       out = (float*)d_out;           // [256,16,32768] f32

    // workspace: bf16 emb_n (32MB) then bf16 dec_n (4MB)
    __bf16* embN = (__bf16*)d_ws;
    __bf16* decn = (__bf16*)((char*)d_ws + (size_t)N_SYM * N_DIMS * sizeof(__bf16));

    emb_norm_kernel<<<N_SYM, 256, 0, stream>>>(emb, embN);
    dec_prep_kernel<<<BATCH, 256, 0, stream>>>(emb, pos, ids, decn);

    dim3 grid(N_SYM / TN, M_TOT / TM);   // (256, 64)
    sim_gemm_kernel<<<grid, 256, 0, stream>>>(decn, embN, out);
}
